// RNNBase_81741817577603
// MI455X (gfx1250) — compile-verified, matrix operand order checked by reading, not run on hardware
//
#include <hip/hip_runtime.h>
#include <hip/hip_bf16.h>
#include <math.h>

// Problem constants (from reference)
#define B_ 32
#define T_ 168
#define M_ 1024
#define A_ 8
#define H_ 64
#define F_ 256   // 4*H

// LDS pitches (padded for bank-conflict-free ds_load_b128 / f32 access)
#define XP   136  // X row pitch in halves (K=128 + pad), 272B = 16B-aligned, +4 banks/row
#define WTP  136  // WT row pitch in halves
#define GP   260  // gates row pitch in floats
#define CP   68   // cell-state row pitch in floats

typedef __attribute__((ext_vector_type(16))) _Float16 v16h;
typedef __attribute__((ext_vector_type(8)))  _Float16 v8h;
typedef __attribute__((ext_vector_type(8)))  float    v8f;

// LDS byte offsets
#define OFF_WT      0                    // [256][WTP] f16 = 69632
#define OFF_X       69632                // [32][XP]   f16 = 8704
#define OFF_GATES   78336                // [32][GP]   f32 = 33280
#define OFF_C       111616               // [32][CP]   f32 = 8704
#define OFF_AW      120320               // [8][64]    f32 = 2048
#define OFF_AB      122368               // [64]       f32 = 256
#define OFF_DW      122624               // [64]       f32 = 256
#define OFF_BIAS    122880               // [256]      f32 = 1024
#define OFF_OPART   123904               // [32][8]    f32 = 1024
#define OFF_MN      124928               // [32]       f32 = 128
#define OFF_SC      125056               // [32]       f32 = 128
#define OFF_SDW     125184               // [1]        f32 (+pad)
#define SMEM_BYTES  125200

// Load one 16x16x32 f16 operand fragment from an LDS row (row-major, K contiguous).
// Per ISA 7.12.2 (16-bit A-matrix 16x32): lanes 0-15 hold K={0..7,16..23} of row M=lane,
// lanes 16-31 hold K={8..15,24..31}. B mirrors with N on lanes (WT stored N-major).
static __device__ inline v16h load_frag16(const _Float16* rowPtr, int kb) {
    union { v16h v; v8h h[2]; } u;
    u.h[0] = *(const v8h*)(rowPtr + kb);        // K = kb+0..7     (VGPRs 0-3)
    u.h[1] = *(const v8h*)(rowPtr + 16 + kb);   // K = 16+kb+0..7  (VGPRs 4-7)
    return u.v;
}

static __device__ inline float sigmoidf_(float x) {
    return 1.f / (1.f + __expf(-x));
}

__global__ void __launch_bounds__(256)
icfa_lstm_kernel(const float* __restrict__ input,
                 const float* __restrict__ assoc_w,
                 const float* __restrict__ assoc_b,
                 const float* __restrict__ w_ih,
                 const float* __restrict__ w_hh,
                 const float* __restrict__ b_ih,
                 const float* __restrict__ b_hh,
                 const float* __restrict__ dense_w,
                 const float* __restrict__ dense_b,
                 float* __restrict__ out)
{
    extern __shared__ char smem[];
    _Float16* WT    = (_Float16*)(smem + OFF_WT);    // [256][WTP] fused [w_ih;w_hh]^T, f16
    _Float16* X     = (_Float16*)(smem + OFF_X);     // [32][XP]  A-matrix: [y | h] f16
    float*    gates = (float*)(smem + OFF_GATES);    // [32][GP]
    float*    cst   = (float*)(smem + OFF_C);        // [32][CP]
    float*    aw    = (float*)(smem + OFF_AW);       // [8][64]
    float*    ab    = (float*)(smem + OFF_AB);       // [64]
    float*    dw    = (float*)(smem + OFF_DW);       // [64]
    float*    bias  = (float*)(smem + OFF_BIAS);     // [256]
    float*    oPart = (float*)(smem + OFF_OPART);    // [32][8]
    float*    mnL   = (float*)(smem + OFF_MN);       // [32]
    float*    scL   = (float*)(smem + OFF_SC);       // [32]
    float*    sdwL  = (float*)(smem + OFF_SDW);      // [1]

    const int m    = blockIdx.x;
    const int tid  = threadIdx.x;
    const int lane = tid & 31;
    const int wv   = tid >> 5;

    // ---------------- one-time per-map-unit init ----------------
    const float* wih = w_ih + (size_t)m * H_ * F_;
    const float* whh = w_hh + (size_t)m * H_ * F_;
    #pragma unroll 4
    for (int idx = tid; idx < H_ * F_; idx += 256) {
        int h = idx >> 8, f = idx & 255;          // idx = h*256 + f -> coalesced global
        WT[f * WTP + h]       = (_Float16)wih[idx];
        WT[f * WTP + H_ + h]  = (_Float16)whh[idx];
    }
    if (tid < F_) bias[tid] = b_ih[m * F_ + tid] + b_hh[m * F_ + tid];
    for (int idx = tid; idx < A_ * H_; idx += 256)
        aw[idx] = assoc_w[(size_t)m * A_ * H_ + idx];
    if (tid < H_) {
        ab[tid] = assoc_b[m * H_ + tid];
        dw[tid] = dense_w[m * H_ + tid];
    }
    for (int idx = tid; idx < B_ * H_; idx += 256) {
        int b = idx >> 6, h = idx & 63;
        X[b * XP + H_ + h] = (_Float16)0.f;       // h0 = 0
        cst[b * CP + h]    = 0.f;                 // c0 = 0
    }
    __syncthreads();
    if (tid == 0) { float s = 0.f; for (int h = 0; h < H_; ++h) s += dw[h]; sdwL[0] = s; }

    const float db = dense_b[m];
    const int b_p   = tid >> 3;          // batch index for P1/P2/P5
    const int a_p   = tid & 7;           // attr index for P1
    const int h0    = (tid & 7) * 8;     // hidden sub-range for P2/P5
    // GEMM tile assignment: 2 M-tiles x 16 N-tiles, 4 tiles/wave
    const int mtile  = wv >> 2;
    const int ntile0 = (wv & 3) * 4;
    const int kb     = (lane < 16) ? 0 : 8;
    const int mrow   = mtile * 16 + (lane & 15);
    const int ncl    = lane & 15;
    const int rbase  = mtile * 16 + ((lane < 16) ? 0 : 8);
    __syncthreads();   // WT / bias / sdw ready

    // Hoist the loop-invariant B-matrix (weight) fragments + biases into registers:
    // 16 fragments x 8 VGPRs = 128 VGPRs held live across the whole time loop.
    v16h bfr[4][4];
    float bvv[4];
    #pragma unroll
    for (int i = 0; i < 4; ++i) {
        int ncol = (ntile0 + i) * 16 + ncl;
        bvv[i] = bias[ncol];
        #pragma unroll
        for (int kt = 0; kt < 4; ++kt)
            bfr[i][kt] = load_frag16(WT + ncol * WTP + kt * 32, kb);
    }

    // ---------------- time loop (sequential recurrence) ----------------
    for (int t = 0; t < T_; ++t) {
        // P1: load x[b, t, m, a], min/max over attrs within 8-lane groups
        float v = input[(((size_t)b_p * T_ + t) * M_ + m) * A_ + a_p];
        float mn = v, mx = v;
        #pragma unroll
        for (int s = 1; s < 8; s <<= 1) {
            mn = fminf(mn, __shfl_xor(mn, s, 8));
            mx = fmaxf(mx, __shfl_xor(mx, s, 8));
        }
        float scale = mx - mn;
        float xnv = (v - mn) / scale;
        float xa[8];
        #pragma unroll
        for (int j = 0; j < 8; ++j) xa[j] = __shfl(xnv, j, 8);
        if (a_p == 0) { mnL[b_p] = mn; scL[b_p] = scale; }

        // P2: association layer y = xn @ aw + ab (VALU), write f16 A-matrix + out partial
        float pp = 0.f;
        #pragma unroll
        for (int j = 0; j < 8; ++j) {
            int h = h0 + j;
            float acc = ab[h];
            #pragma unroll
            for (int a = 0; a < 8; ++a) acc += xa[a] * aw[a * H_ + h];
            X[b_p * XP + h] = (_Float16)acc;
            pp += acc * dw[h];
        }
        oPart[b_p * 8 + a_p] = pp;
        __syncthreads();   // barrier A: X(y) + oPart + mn/sc ready

        // out[b,t,m] = scale*sum(y*dw) + mn*sum(dw) + db  (wave 0 only)
        if (tid < 32) {
            float s = 0.f;
            #pragma unroll
            for (int j = 0; j < 8; ++j) s += oPart[tid * 8 + j];
            out[((size_t)tid * T_ + t) * M_ + m] = scL[tid] * s + mnL[tid] * sdwL[0] + db;
        }

        // P3: gates = [y|h] @ [w_ih;w_hh] + bias via WMMA f16, K=128
        // B operands are register-resident; only A fragments stream from LDS.
        v16h afr[4];
        #pragma unroll
        for (int kt = 0; kt < 4; ++kt)
            afr[kt] = load_frag16(X + mrow * XP + kt * 32, kb);
        #pragma unroll
        for (int i = 0; i < 4; ++i) {
            int ncol = (ntile0 + i) * 16 + ncl;
            v8f acc = {};
            #pragma unroll
            for (int kt = 0; kt < 4; ++kt) {
                acc = __builtin_amdgcn_wmma_f32_16x16x32_f16(
                        false, afr[kt], false, bfr[i][kt], (short)0, acc, false, false);
            }
            float bv = bvv[i];
            #pragma unroll
            for (int r = 0; r < 8; ++r)
                gates[(rbase + r) * GP + ncol] = acc[r] + bv;
        }
        __syncthreads();   // barrier B: gates ready

        // P5: LSTM pointwise update; h -> X (f16) for next step's A matrix
        #pragma unroll
        for (int j = 0; j < 8; ++j) {
            int h = h0 + j;
            float gi = gates[b_p * GP + h];
            float gf = gates[b_p * GP + 64 + h];
            float gg = gates[b_p * GP + 128 + h];
            float go = gates[b_p * GP + 192 + h];
            float cn = sigmoidf_(gf) * cst[b_p * CP + h] + sigmoidf_(gi) * tanhf(gg);
            cst[b_p * CP + h] = cn;
            X[b_p * XP + H_ + h] = (_Float16)(sigmoidf_(go) * tanhf(cn));
        }
        // next iteration's barrier A separates these writes from the next GEMM reads
    }
}

extern "C" void kernel_launch(void* const* d_in, const int* in_sizes, int n_in,
                              void* d_out, int out_size, void* d_ws, size_t ws_size,
                              hipStream_t stream) {
    const float* input   = (const float*)d_in[0];
    const float* assoc_w = (const float*)d_in[1];
    const float* assoc_b = (const float*)d_in[2];
    const float* w_ih    = (const float*)d_in[3];
    const float* w_hh    = (const float*)d_in[4];
    const float* b_ih    = (const float*)d_in[5];
    const float* b_hh    = (const float*)d_in[6];
    const float* dense_w = (const float*)d_in[7];
    const float* dense_b = (const float*)d_in[8];
    float* out = (float*)d_out;

    (void)in_sizes; (void)n_in; (void)out_size; (void)d_ws; (void)ws_size;

    hipFuncSetAttribute((const void*)icfa_lstm_kernel,
                        hipFuncAttributeMaxDynamicSharedMemorySize, SMEM_BYTES);

    dim3 grid(M_);
    dim3 block(256);
    icfa_lstm_kernel<<<grid, block, SMEM_BYTES, stream>>>(
        input, assoc_w, assoc_b, w_ih, w_hh, b_ih, b_hh, dense_w, dense_b, out);
}
